// WaveQuantLinear_23725399343425
// MI455X (gfx1250) — 8-bit matrix operand order repaired and checked by reading, compiled, not run on hardware
//
#include <hip/hip_runtime.h>

#define INPUT_SCALE  0.05f
#define WEIGHT_SCALE 0.02f
#define FP8_CLAMP    240.0f

#define K_DIM 4096
#define M_TOT 8192
#define N_DIM 4096
#define KSTEPS (K_DIM / 128)
#define TILE_BYTES 2048
#define RBLK_BYTES (KSTEPS * TILE_BYTES)
#define LDS_BUF_BYTES (16 * TILE_BYTES)

typedef __attribute__((ext_vector_type(16))) int      v16i;
typedef __attribute__((ext_vector_type(8)))  float    v8f;
typedef __attribute__((ext_vector_type(4)))  int      v4i;
typedef __attribute__((ext_vector_type(4)))  float    v4f;
typedef __attribute__((ext_vector_type(2)))  unsigned v2u;

__device__ __forceinline__ unsigned char f32_to_fp8_sw(float x) {
  union { float f; unsigned u; } c; c.f = x;
  unsigned sign = (c.u >> 31) & 1u;
  int      exp  = (int)((c.u >> 23) & 0xFF) - 127;
  unsigned mant = c.u & 0x7FFFFFu;
  int e8 = exp + 8;
  if (e8 <= 0) return (unsigned char)(sign << 7);
  unsigned m3 = (mant + 0x80000u) >> 20;
  if (m3 == 8u) { m3 = 0u; e8++; }
  if (e8 > 15) { e8 = 15; m3 = 7u; }
  return (unsigned char)((sign << 7) | ((unsigned)e8 << 3) | m3);
}

__device__ __forceinline__ unsigned pack4_fp8(float a, float b, float c, float d) {
#if defined(__gfx1250__) && __has_builtin(__builtin_amdgcn_cvt_pk_fp8_f32)
  int r = __builtin_amdgcn_cvt_pk_fp8_f32(a, b, 0, false);
  r = __builtin_amdgcn_cvt_pk_fp8_f32(c, d, r, true);
  return (unsigned)r;
#else
  return (unsigned)f32_to_fp8_sw(a)        |
         ((unsigned)f32_to_fp8_sw(b) << 8) |
         ((unsigned)f32_to_fp8_sw(c) << 16)|
         ((unsigned)f32_to_fp8_sw(d) << 24);
#endif
}

__device__ __forceinline__ float scale_clamp(float v, float inv_scale) {
  return fminf(fmaxf(v * inv_scale, -FP8_CLAMP), FP8_CLAMP);
}

template <bool BSWZ>
__global__ __launch_bounds__(256) void quant_swizzle_kernel(
    const float* __restrict__ in, unsigned char* __restrict__ out,
    float inv_scale, long n_groups) {
  long idx    = (long)blockIdx.x * 256 + threadIdx.x;
  long stride = (long)gridDim.x * 256;
  for (; idx < n_groups; idx += stride) {
    const v4f* p = (const v4f*)(in + idx * 8);
    v4f lo = __builtin_nontemporal_load(p);
    v4f hi = __builtin_nontemporal_load(p + 1);
    v2u r;
    r.x = pack4_fp8(scale_clamp(lo[0], inv_scale), scale_clamp(lo[1], inv_scale),
                    scale_clamp(lo[2], inv_scale), scale_clamp(lo[3], inv_scale));
    r.y = pack4_fp8(scale_clamp(hi[0], inv_scale), scale_clamp(hi[1], inv_scale),
                    scale_clamp(hi[2], inv_scale), scale_clamp(hi[3], inv_scale));

    long base = idx * 8;
    long row  = base >> 12;
    int  kb   = (int)(base & (K_DIM - 1));
    int  kblk = kb >> 7;
    int  kl   = kb & 127;
    long tile = (row >> 4) * (long)KSTEPS + kblk;
    int lane, boff;
    if (BSWZ) {
      int h = (kl >> 4) & 1;
      lane = (int)(row & 15) + 16 * h;
      boff = (kl >> 5) * 512 + lane * 16 + (kl & 15);
    } else {
      int h  = (kl >> 3) & 1;
      int p8 = ((kl & 63) >> 4) + 4 * (kl >> 6);
      lane = (int)(row & 15) + 16 * h;
      boff = (p8 >> 1) * 512 + lane * 16 + (p8 & 1) * 8;
    }
    __builtin_nontemporal_store(r, (v2u*)(out + tile * TILE_BYTES + boff));
  }
}

__device__ __forceinline__ void async_copy_128B(unsigned lds_off,
                                                const unsigned char* g) {
  unsigned long long ga = (unsigned long long)g;
  asm volatile(
      "global_load_async_to_lds_b128 %0, %1, off\n\t"
      "global_load_async_to_lds_b128 %0, %1, off offset:16\n\t"
      "global_load_async_to_lds_b128 %0, %1, off offset:32\n\t"
      "global_load_async_to_lds_b128 %0, %1, off offset:48\n\t"
      "global_load_async_to_lds_b128 %0, %1, off offset:64\n\t"
      "global_load_async_to_lds_b128 %0, %1, off offset:80\n\t"
      "global_load_async_to_lds_b128 %0, %1, off offset:96\n\t"
      "global_load_async_to_lds_b128 %0, %1, off offset:112"
      :: "v"(lds_off), "v"(ga) : "memory");
}

__device__ __forceinline__ v16i lds_frag(const unsigned char* tile, int lane) {
  v16i f;
#pragma unroll
  for (int q = 0; q < 4; ++q) {
    v4i d = *(const v4i*)(tile + q * 512 + lane * 16);
    f[4 * q + 0] = d[0];
    f[4 * q + 1] = d[1];
    f[4 * q + 2] = d[2];
    f[4 * q + 3] = d[3];
  }
  return f;
}

__device__ __forceinline__ void compute_step(const unsigned char* bufp, int lane,
                                             int waveM, int waveN, v8f acc[2][4]) {
  v16i a0 = lds_frag(bufp + (waveM * 2 + 0) * TILE_BYTES, lane);
  v16i a1 = lds_frag(bufp + (waveM * 2 + 1) * TILE_BYTES, lane);
  v16i bc = lds_frag(bufp + (8 + waveN * 4 + 0) * TILE_BYTES, lane);
#pragma unroll
  for (int j = 0; j < 4; ++j) {
    v16i bn;
    if (j < 3) bn = lds_frag(bufp + (8 + waveN * 4 + j + 1) * TILE_BYTES, lane);
    acc[0][j] = __builtin_amdgcn_wmma_f32_16x16x128_fp8_fp8(
        a0, bc, (short)0, acc[0][j], false, false);
    acc[1][j] = __builtin_amdgcn_wmma_f32_16x16x128_fp8_fp8(
        a1, bc, (short)0, acc[1][j], false, false);
    if (j < 3) bc = bn;
  }
}

__global__ __launch_bounds__(256) void fp8_gemm_wmma_kernel(
    const unsigned char* __restrict__ Aq,
    const unsigned char* __restrict__ Bq,
    float* __restrict__ out) {
  __shared__ unsigned char smem[4][16][TILE_BYTES];

  const int tid  = threadIdx.x;
  const int lane = tid & 31;
  const int wave = tid >> 5;
  const int half = lane >> 4;
  const int l15  = lane & 15;
  const int waveM = wave >> 1;
  const int waveN = wave & 1;

  const long tileM = (long)blockIdx.y * 128 + (long)waveM * 32;
  const long tileN = (long)blockIdx.x * 128 + (long)waveN * 64;

  const int ctile = tid >> 4;
  const int csub  = tid & 15;
  const unsigned char* gsrc =
      (ctile < 8)
          ? Aq + ((long)blockIdx.y * 8 + ctile) * (long)RBLK_BYTES + csub * 128
          : Bq + ((long)blockIdx.x * 8 + (ctile - 8)) * (long)RBLK_BYTES + csub * 128;
  const unsigned lds_slot  = (unsigned)(ctile * TILE_BYTES + csub * 128);
  const unsigned smem_base = (unsigned)(unsigned long long)(uintptr_t)&smem[0][0][0];

  async_copy_128B(smem_base + lds_slot, gsrc);

  v8f acc[2][4] = {};

#pragma unroll 4
  for (int ks = 0; ks < KSTEPS - 4; ++ks) {
    async_copy_128B(
        smem_base + (unsigned)(((ks + 1) & 3) * LDS_BUF_BYTES) + lds_slot,
        gsrc + (long)(ks + 1) * TILE_BYTES);
    asm volatile("s_wait_asynccnt 0x8" ::: "memory");
    __syncthreads();
    compute_step(&smem[ks & 3][0][0], lane, waveM, waveN, acc);
  }

#pragma unroll
  for (int ks = KSTEPS - 4; ks < KSTEPS; ++ks) {
    if (ks + 1 < KSTEPS) {
      async_copy_128B(
          smem_base + (unsigned)(((ks + 1) & 3) * LDS_BUF_BYTES) + lds_slot,
          gsrc + (long)(ks + 1) * TILE_BYTES);
      asm volatile("s_wait_asynccnt 0x8" ::: "memory");
    } else {
      asm volatile("s_wait_asynccnt 0x0" ::: "memory");
    }
    __syncthreads();
    compute_step(&smem[ks & 3][0][0], lane, waveM, waveN, acc);
  }

  const float dq = INPUT_SCALE * WEIGHT_SCALE;
#pragma unroll
  for (int i = 0; i < 2; ++i) {
#pragma unroll
    for (int r = 0; r < 8; ++r) {
      long row = tileM + i * 16 + (long)half * 8 + r;
      float* orow = out + row * N_DIM + tileN + l15;
#pragma unroll
      for (int j = 0; j < 4; ++j)
        __builtin_nontemporal_store(acc[i][j][r] * dq, orow + j * 16);
    }
  }
}

extern "C" void kernel_launch(void* const* d_in, const int* in_sizes, int n_in,
                              void* d_out, int out_size, void* d_ws, size_t ws_size,
                              hipStream_t stream) {
  const float* x = (const float*)d_in[0];
  const float* w = (const float*)d_in[1];
  float* out = (float*)d_out;

  unsigned char* Aq = (unsigned char*)d_ws;
  unsigned char* Bq = Aq + (size_t)M_TOT * K_DIM;

  const long a_groups = (long)M_TOT * K_DIM / 8;
  const long b_groups = (long)N_DIM * K_DIM / 8;
  quant_swizzle_kernel<true ><<<2048, 256, 0, stream>>>(x, Aq, 1.0f / INPUT_SCALE,  a_groups);
  quant_swizzle_kernel<true ><<<1024, 256, 0, stream>>>(w, Bq, 1.0f / WEIGHT_SCALE, b_groups);

  dim3 grid(N_DIM / 128, M_TOT / 128);
  fp8_gemm_wmma_kernel<<<grid, 256, 0, stream>>>(Aq, Bq, out);
}
